// my_gpt_77567109366242
// MI455X (gfx1250) — matrix operand order repaired and checked by reading, not compiled, not run on hardware
//
#include <hip/hip_runtime.h>

// ---------------------------------------------------------------- types
typedef __attribute__((ext_vector_type(16))) _Float16 v16h;
typedef __attribute__((ext_vector_type(8)))  float    v8f;

union Frag { v16h v; unsigned u[8]; };

static __device__ __forceinline__ _Float16 f2h(float f) { return (_Float16)f; }

// ---------------------------------------------------------------- consts
#define NL 24
#define NH 32
#define NC 512
#define NHS 16
#define NV 65
#define NVP 128            // LM head padded cols (so B tiles always in-bounds)
#define NB 32
#define NT 256
#define NM (NB * NT)       // 8192 tokens

// ================================================================ GEMM
// C[M,Nout] = A[M,K] (f16) * B[K,N] (f16), f32 accum via WMMA.
// N = B leading dim (mult of 64 here); Nout = output leading dim + store guard.
// Tile: 64x64 per block, K-stage 64 (2 WMMA chunks), 8 waves, wave = 16x32.
// Tiles staged with GLOBAL_LOAD_ASYNC_TO_LDS_B128 (ASYNCcnt).
__global__ __launch_bounds__(256) void gpt_gemm_wmma(
    const _Float16* __restrict__ A, const _Float16* __restrict__ Bw,
    float* outF, _Float16* outH,
    const float* __restrict__ bias, const float* res,
    int M, int N, int K, int relu, int Nout)
{
    __shared__ _Float16 As[64][72];   // 64x64 tile, +8 pad (row = 144B, 16B aligned)
    __shared__ _Float16 Bs[64][72];

    const int tid  = threadIdx.x;
    const int lane = tid & 31;
    const int wave = tid >> 5;          // 0..7
    const int wm   = wave >> 1;         // 0..3  (M sub-tile)
    const int wn   = wave & 1;          // 0..1  (N sub-tile pair)
    const int m0   = blockIdx.x * 64;
    const int n0   = blockIdx.y * 64;

    v8f acc0 = {0.f,0.f,0.f,0.f,0.f,0.f,0.f,0.f};
    v8f acc1 = {0.f,0.f,0.f,0.f,0.f,0.f,0.f,0.f};

    // staging role: each thread owns 16 halfs of A and 16 of B (2 x B128 each)
    const int sr = tid >> 2;            // 0..63  (row of A tile / row of B tile)
    const int sc = (tid & 3) * 16;      // 0,16,32,48
    const unsigned ldsA = (unsigned)(uintptr_t)&As[sr][sc];
    const unsigned ldsB = (unsigned)(uintptr_t)&Bs[sr][sc];

    for (int kt = 0; kt < K; kt += 64) {
        const _Float16* ga = A  + (size_t)(m0 + sr) * K + kt + sc;
        const _Float16* gb = Bw + (size_t)(kt + sr) * N + n0 + sc;
        // async DMA global -> LDS (inst offset applies to both addresses)
        asm volatile(
            "global_load_async_to_lds_b128 %0, %2, off\n\t"
            "global_load_async_to_lds_b128 %0, %2, off offset:16\n\t"
            "global_load_async_to_lds_b128 %1, %3, off\n\t"
            "global_load_async_to_lds_b128 %1, %3, off offset:16"
            :: "v"(ldsA), "v"(ldsB), "v"(ga), "v"(gb) : "memory");
        asm volatile("s_wait_asynccnt 0" ::: "memory");
        __syncthreads();

        const int m = lane & 15, g = lane >> 4;
        #pragma unroll
        for (int kk = 0; kk < 64; kk += 32) {
            Frag a;
            #pragma unroll
            for (int r = 0; r < 8; ++r) {
                int kb = kk + 2*r + ((r >= 4) ? 8 : 0) + g*8;
                a.u[r] = *(const unsigned*)&As[wm*16 + m][kb];
            }
            Frag b0, b1;
            const int nb = wn * 32;
            #pragma unroll
            for (int r = 0; r < 8; ++r) {
                int kb = kk + 2*r + ((r >= 4) ? 8 : 0) + g*8;
                b0.v[2*r]     = Bs[kb    ][nb +      m];
                b0.v[2*r + 1] = Bs[kb + 1][nb +      m];
                b1.v[2*r]     = Bs[kb    ][nb + 16 + m];
                b1.v[2*r + 1] = Bs[kb + 1][nb + 16 + m];
            }
            acc0 = __builtin_amdgcn_wmma_f32_16x16x32_f16(false, a.v, false, b0.v,
                                                          (short)0, acc0, false, false);
            acc1 = __builtin_amdgcn_wmma_f32_16x16x32_f16(false, a.v, false, b1.v,
                                                          (short)0, acc1, false, false);
        }
        __syncthreads();
    }

    // ---- epilogue
    const int g  = lane >> 4;
    const int nn = lane & 15;
    #pragma unroll
    for (int r = 0; r < 8; ++r) {
        int mo  = r + 8*g;
        int row = m0 + wm*16 + mo;
        #pragma unroll
        for (int t = 0; t < 2; ++t) {
            int col = n0 + wn*32 + t*16 + nn;
            if (col >= Nout) continue;
            float v = (t == 0) ? acc0[r] : acc1[r];
            if (bias) v += bias[col];
            if (outH) {
                if (relu) v = fmaxf(v, 0.0f);
                outH[(size_t)row * Nout + col] = f2h(v);
            } else {
                if (res) v += res[(size_t)row * Nout + col];
                outF[(size_t)row * Nout + col] = v;
            }
        }
    }
}

// ================================================================ fused attention
// One block = (query block of 64 rows) x (head h) x (batch b).
// qkv: [B*T, 1536] f16 (q|k|v each 512 = 32 heads * 16). ao: [B*T, 512] f16.
__global__ __launch_bounds__(256) void gpt_attn_wmma(
    const _Float16* __restrict__ qkv, _Float16* __restrict__ ao)
{
    __shared__ _Float16 Qs[64][34];    // 64 x 32 (HS padded to 32 with zeros)
    __shared__ _Float16 KT[32][258];   // K transposed: [d(pad 32)][s=256]
    __shared__ _Float16 Vs[256][18];   // [s][d]
    __shared__ float    S[64][260];    // scores; reused as f16 P in place

    const int tid  = threadIdx.x;
    const int lane = tid & 31;
    const int wave = tid >> 5;
    const int qb = blockIdx.x, h = blockIdx.y, b = blockIdx.z;
    const int tq0 = qb * 64;
    const size_t base = (size_t)b * NT * 1536;

    // stage Q (padded), K^T (padded), V
    for (int i = tid; i < 64 * 32; i += 256) {
        int r = i >> 5, c = i & 31;
        Qs[r][c] = (c < NHS)
            ? qkv[base + (size_t)(tq0 + r) * 1536 + h * NHS + c] : (_Float16)0.0f;
    }
    for (int i = tid; i < 32 * 256; i += 256) {
        int d = i >> 8, s = i & 255;
        KT[d][s] = (d < NHS)
            ? qkv[base + (size_t)s * 1536 + 512 + h * NHS + d] : (_Float16)0.0f;
    }
    for (int i = tid; i < 256 * 16; i += 256) {
        int s = i >> 4, d = i & 15;
        Vs[s][d] = qkv[base + (size_t)s * 1536 + 1024 + h * NHS + d];
    }
    __syncthreads();

    // ---- S = (Q K^T) * HS^-0.5, causal mask.  8 waves x 8 tiles.
    {
        const int mt = wave >> 1;
        const int m = lane & 15, g = lane >> 4;
        Frag a;
        #pragma unroll
        for (int r = 0; r < 8; ++r) {
            int kb = 2*r + ((r >= 4) ? 8 : 0) + g*8;
            a.u[r] = *(const unsigned*)&Qs[mt*16 + m][kb];
        }
        for (int i = 0; i < 8; ++i) {
            int nt = (wave & 1) * 8 + i;
            Frag bf;
            #pragma unroll
            for (int r = 0; r < 8; ++r) {
                int kb = 2*r + ((r >= 4) ? 8 : 0) + g*8;
                bf.v[2*r]     = KT[kb    ][nt*16 + m];
                bf.v[2*r + 1] = KT[kb + 1][nt*16 + m];
            }
            v8f acc = {0.f,0.f,0.f,0.f,0.f,0.f,0.f,0.f};
            acc = __builtin_amdgcn_wmma_f32_16x16x32_f16(false, a.v, false, bf.v,
                                                         (short)0, acc, false, false);
            const int nn = lane & 15;
            #pragma unroll
            for (int r = 0; r < 8; ++r) {
                int mo   = r + 8*g;
                int qrow = tq0 + mt*16 + mo;
                int col  = nt*16 + nn;
                float v = acc[r] * 0.25f;   // HS^-0.5, HS=16
                S[mt*16 + mo][col] = (col <= qrow) ? v : -3.0e38f;
            }
        }
    }
    __syncthreads();

    // ---- softmax per row; write f16 P in place over the S rows
    if (tid < 64) {
        int r = tid;
        float mx = -3.0e38f;
        for (int s = 0; s < 256; ++s) mx = fmaxf(mx, S[r][s]);
        float sum = 0.0f;
        for (int s = 0; s < 256; ++s) { float e = __expf(S[r][s] - mx); S[r][s] = e; sum += e; }
        float inv = 1.0f / sum;
        _Float16* P = (_Float16*)&S[r][0];
        for (int s = 0; s < 256; ++s) P[s] = f2h(S[r][s] * inv);  // safe: s/2 <= s
    }
    __syncthreads();

    // ---- O = P V   (waves 0..3, one 16-row tile each, K=256 in 8 steps)
    if (wave < 4) {
        const int m = lane & 15, g = lane >> 4;
        const _Float16* Pb = (const _Float16*)&S[0][0];
        const int pstride = 260 * 2;    // f16 elements per S row
        v8f acc = {0.f,0.f,0.f,0.f,0.f,0.f,0.f,0.f};
        for (int ks = 0; ks < 256; ks += 32) {
            Frag a, bf;
            #pragma unroll
            for (int r = 0; r < 8; ++r) {
                int kb = ks + 2*r + ((r >= 4) ? 8 : 0) + g*8;
                a.u[r] = *(const unsigned*)&Pb[(size_t)(wave*16 + m) * pstride + kb];
            }
            #pragma unroll
            for (int r = 0; r < 8; ++r) {
                int kb = ks + 2*r + ((r >= 4) ? 8 : 0) + g*8;
                bf.v[2*r]     = Vs[kb    ][m];
                bf.v[2*r + 1] = Vs[kb + 1][m];
            }
            acc = __builtin_amdgcn_wmma_f32_16x16x32_f16(false, a.v, false, bf.v,
                                                         (short)0, acc, false, false);
        }
        const int nn = lane & 15;
        #pragma unroll
        for (int r = 0; r < 8; ++r) {
            int mo   = r + 8*g;
            int trow = tq0 + wave*16 + mo;
            ao[((size_t)b * NT + trow) * NC + h * NHS + nn] = f2h(acc[r]);
        }
    }
}

// ================================================================ LayerNorm (row of 512) -> f16
__global__ __launch_bounds__(256) void gpt_ln(
    const float* __restrict__ x, const float* __restrict__ sc,
    const float* __restrict__ bi, _Float16* __restrict__ out)
{
    __shared__ float r1[256], r2[256];
    const int row = blockIdx.x, tid = threadIdx.x;
    float v0 = x[(size_t)row * NC + tid];
    float v1 = x[(size_t)row * NC + 256 + tid];
    r1[tid] = v0 + v1;
    r2[tid] = v0 * v0 + v1 * v1;
    __syncthreads();
    for (int s = 128; s > 0; s >>= 1) {
        if (tid < s) { r1[tid] += r1[tid + s]; r2[tid] += r2[tid + s]; }
        __syncthreads();
    }
    float mean = r1[0] * (1.0f / NC);
    float var  = r2[0] * (1.0f / NC) - mean * mean;
    float rstd = rsqrtf(var + 1e-5f);
    out[(size_t)row * NC + tid]       = f2h((v0 - mean) * rstd * sc[tid]       + bi[tid]);
    out[(size_t)row * NC + 256 + tid] = f2h((v1 - mean) * rstd * sc[256 + tid] + bi[256 + tid]);
}

// ================================================================ embedding (tok+pos, both indexed by token id)
__global__ __launch_bounds__(256) void gpt_embed(
    const int* __restrict__ idx, const float* __restrict__ tok,
    const float* __restrict__ pos, float* __restrict__ x)
{
    const int row = blockIdx.x, tid = threadIdx.x;
    const int id = idx[row];
    x[(size_t)row * NC + tid]       = tok[(size_t)id * NC + tid]       + pos[(size_t)id * NC + tid];
    x[(size_t)row * NC + 256 + tid] = tok[(size_t)id * NC + 256 + tid] + pos[(size_t)id * NC + 256 + tid];
}

// ================================================================ converters
__global__ void gpt_cvt_f16(const float* __restrict__ in, _Float16* __restrict__ out, int n)
{
    int i = blockIdx.x * 256 + threadIdx.x;
    if (i < n) out[i] = f2h(in[i]);
}

// Wq/Wk/Wv [L,H,C,HS] -> Wqkv [L][C][1536] with col = which*512 + h*16 + d
__global__ void gpt_cvt_qkv(const float* __restrict__ Wq, const float* __restrict__ Wk,
                            const float* __restrict__ Wv, _Float16* __restrict__ out, int total)
{
    int i = blockIdx.x * 256 + threadIdx.x;
    if (i >= total) return;
    int l   = i / (NC * 1536);
    int rem = i % (NC * 1536);
    int c = rem / 1536, j = rem % 1536;
    int which = j / 512, jj = j % 512;
    int hh = jj / NHS, d = jj % NHS;
    const float* W = (which == 0) ? Wq : ((which == 1) ? Wk : Wv);
    out[i] = f2h(W[(((size_t)l * NH + hh) * NC + c) * NHS + d]);
}

// Wlm [C,V] -> padded [C, NVP] f16 (zero beyond V)
__global__ void gpt_cvt_lm(const float* __restrict__ in, _Float16* __restrict__ out, int total)
{
    int i = blockIdx.x * 256 + threadIdx.x;
    if (i >= total) return;
    int c = i / NVP, v = i % NVP;
    out[i] = (v < NV) ? f2h(in[(size_t)c * NV + v]) : (_Float16)0.0f;
}

// ================================================================ host
extern "C" void kernel_launch(void* const* d_in, const int* in_sizes, int n_in,
                              void* d_out, int out_size, void* d_ws, size_t ws_size,
                              hipStream_t stream)
{
    (void)in_sizes; (void)n_in; (void)out_size; (void)ws_size;
    const int*   idx   = (const int*)  d_in[0];
    const float* tok   = (const float*)d_in[1];
    const float* pos   = (const float*)d_in[2];
    const float* Wq    = (const float*)d_in[3];
    const float* Wk    = (const float*)d_in[4];
    const float* Wv    = (const float*)d_in[5];
    const float* Wp    = (const float*)d_in[6];
    const float* bp    = (const float*)d_in[7];
    const float* ln1s  = (const float*)d_in[8];
    const float* ln1b  = (const float*)d_in[9];
    const float* ln2s  = (const float*)d_in[10];
    const float* ln2b  = (const float*)d_in[11];
    const float* W1    = (const float*)d_in[12];
    const float* b1    = (const float*)d_in[13];
    const float* W2    = (const float*)d_in[14];
    const float* b2    = (const float*)d_in[15];
    const float* Wlm   = (const float*)d_in[16];
    const float* blm   = (const float*)d_in[17];

    char* ws = (char*)d_ws;
    size_t off = 0;
    auto take = [&](size_t bytes) { char* p = ws + off; off = (off + bytes + 255) & ~(size_t)255; return p; };

    _Float16* wqkv = (_Float16*)take((size_t)NL * NC * 1536 * 2);
    _Float16* wp   = (_Float16*)take((size_t)NL * NC * NC * 2);
    _Float16* w1   = (_Float16*)take((size_t)NL * NC * 4 * NC * 2);
    _Float16* w2   = (_Float16*)take((size_t)NL * 4 * NC * NC * 2);
    _Float16* wlm  = (_Float16*)take((size_t)NC * NVP * 2);
    float*    x    = (float*)   take((size_t)NM * NC * 4);
    _Float16* xn   = (_Float16*)take((size_t)NM * NC * 2);
    _Float16* qkv  = (_Float16*)take((size_t)NM * 1536 * 2);
    _Float16* aoh  = (_Float16*)take((size_t)NM * NC * 2);
    _Float16* hbuf = (_Float16*)take((size_t)NM * 4 * NC * 2);

    // ---- weight conversion (every call; deterministic)
    {
        int t = NL * NC * 1536;
        gpt_cvt_qkv<<<(t + 255) / 256, 256, 0, stream>>>(Wq, Wk, Wv, wqkv, t);
        t = NL * NC * NC;
        gpt_cvt_f16<<<(t + 255) / 256, 256, 0, stream>>>(Wp, wp, t);
        t = NL * NC * 4 * NC;
        gpt_cvt_f16<<<(t + 255) / 256, 256, 0, stream>>>(W1, w1, t);
        gpt_cvt_f16<<<(t + 255) / 256, 256, 0, stream>>>(W2, w2, t);
        t = NC * NVP;
        gpt_cvt_lm<<<(t + 255) / 256, 256, 0, stream>>>(Wlm, wlm, t);
    }

    // ---- embedding
    gpt_embed<<<NM, 256, 0, stream>>>(idx, tok, pos, x);

    const dim3 blk(256);
    for (int l = 0; l < NL; ++l) {
        const _Float16* wqkv_l = wqkv + (size_t)l * NC * 1536;
        const _Float16* wp_l   = wp   + (size_t)l * NC * NC;
        const _Float16* w1_l   = w1   + (size_t)l * NC * 4 * NC;
        const _Float16* w2_l   = w2   + (size_t)l * 4 * NC * NC;

        gpt_ln<<<NM, blk, 0, stream>>>(x, ln1s + l * NC, ln1b + l * NC, xn);

        // QKV: [8192,512] x [512,1536] -> f16
        gpt_gemm_wmma<<<dim3(NM / 64, 1536 / 64), blk, 0, stream>>>(
            xn, wqkv_l, nullptr, qkv, nullptr, nullptr, NM, 1536, NC, 0, 1536);

        gpt_attn_wmma<<<dim3(NT / 64, NH, NB), blk, 0, stream>>>(qkv, aoh);

        // proj + residual + bias -> x (f32)
        gpt_gemm_wmma<<<dim3(NM / 64, NC / 64), blk, 0, stream>>>(
            aoh, wp_l, x, nullptr, bp + l * NC, x, NM, NC, NC, 0, NC);

        gpt_ln<<<NM, blk, 0, stream>>>(x, ln2s + l * NC, ln2b + l * NC, xn);

        // MLP1: bias + ReLU -> f16 hidden
        gpt_gemm_wmma<<<dim3(NM / 64, (4 * NC) / 64), blk, 0, stream>>>(
            xn, w1_l, nullptr, hbuf, b1 + l * 4 * NC, nullptr, NM, 4 * NC, NC, 1, 4 * NC);

        // MLP2: bias + residual -> x (f32)
        gpt_gemm_wmma<<<dim3(NM / 64, NC / 64), blk, 0, stream>>>(
            hbuf, w2_l, x, nullptr, b2 + l * NC, x, NM, NC, 4 * NC, 0, NC);
    }

    // ---- LM head (B padded to NVP cols; store guarded to 65)
    gpt_cvt_f16<<<(NM * NC + 255) / 256, 256, 0, stream>>>(x, xn, NM * NC);
    gpt_gemm_wmma<<<dim3(NM / 64, NVP / 64), blk, 0, stream>>>(
        xn, wlm, (float*)d_out, nullptr, blm, nullptr, NM, NVP, NC, 0, NV);
}
